// DistNet_28080496181610
// MI455X (gfx1250) — matrix operand order repaired and checked by reading, not compiled, not run on hardware
//
#include <hip/hip_runtime.h>
#include <hip/hip_bf16.h>
#include <math.h>

// ---------------------------------------------------------------------------
// Types for WMMA
// ---------------------------------------------------------------------------
typedef __attribute__((ext_vector_type(16))) _Float16 v16h;
typedef __attribute__((ext_vector_type(8)))  _Float16 v8h;
typedef __attribute__((ext_vector_type(8)))  float    v8f;

#define FLAG_GELU 1
#define FLAG_ACC  2

#define TM 128
#define TN 128
#define TKK 32
#define LDS_STRIDE 40   // halves per row (32 data + 8 pad), 80B row stride (16B aligned)

// ---------------------------------------------------------------------------
// WMMA fragment loaders from LDS (per ISA 16-bit A/B VGPR layouts, wave32)
// A: elements 0..7 -> K = 8*g..8*g+7 ; elements 8..15 -> K = 16+8*g..16+8*g+7
// B: elements 0..7 -> K = 8*g..8*g+7 ; elements 8..15 -> K = 16+8*g..16+8*g+7
// ---------------------------------------------------------------------------
__device__ inline v16h load_fragA(const _Float16* rowp, int g) {
  union { v16h v; v8h h[2]; } u;
  u.h[0] = *(const v8h*)(rowp + 8 * g);
  u.h[1] = *(const v8h*)(rowp + 16 + 8 * g);
  return u.v;
}
__device__ inline v16h load_fragB(const _Float16* rowp, int g) {
  union { v16h v; v8h h[2]; } u;
  u.h[0] = *(const v8h*)(rowp + 8 * g);
  u.h[1] = *(const v8h*)(rowp + 16 + 8 * g);
  return u.v;
}

// Load 16 consecutive f32, convert to f16, return as two v8h.
__device__ inline void load_cvt16(const float* __restrict__ src, v8h out[2]) {
  const float4* p = (const float4*)src;
  float4 f0 = p[0], f1 = p[1], f2 = p[2], f3 = p[3];
  union { v8h v; _Float16 e[8]; } a, b;
  a.e[0]=(_Float16)f0.x; a.e[1]=(_Float16)f0.y; a.e[2]=(_Float16)f0.z; a.e[3]=(_Float16)f0.w;
  a.e[4]=(_Float16)f1.x; a.e[5]=(_Float16)f1.y; a.e[6]=(_Float16)f1.z; a.e[7]=(_Float16)f1.w;
  b.e[0]=(_Float16)f2.x; b.e[1]=(_Float16)f2.y; b.e[2]=(_Float16)f2.z; b.e[3]=(_Float16)f2.w;
  b.e[4]=(_Float16)f3.x; b.e[5]=(_Float16)f3.y; b.e[6]=(_Float16)f3.z; b.e[7]=(_Float16)f3.w;
  out[0] = a.v;
  out[1] = b.v;
}

__device__ inline void store16_lds(_Float16* dst, const v8h src[2]) {
  *(v8h*)dst       = src[0];
  *(v8h*)(dst + 8) = src[1];
}

// ---------------------------------------------------------------------------
// Fused WMMA GEMM:  C[m,n] = sum_k A[rowmap(m), k] * W[n, k]  (+bias+gelu+resid)
// A: f32 (M x K row-major, with optional circular row shift per batch of Lrow
//    rows);  W: f32 (N x K row-major);  C: f32 (M x N).
// REQUIREMENTS (guaranteed by the launcher): M % 128 == 0, N % 128 == 0,
// K % 32 == 0 — no edge guards, branch-free inner loop + epilogue.
// Double-buffered LDS: one barrier per K-step; next tile's global loads +
// f16 conversion overlap the current tile's WMMAs.
// ---------------------------------------------------------------------------
__global__ __launch_bounds__(256)
void gemm_f16wmma(const float* __restrict__ A, const float* __restrict__ W,
                  const float* __restrict__ bias, const float* __restrict__ resid,
                  float* __restrict__ C,
                  int M, int N, int K, int Lrow, int shift, int flags)
{
  (void)M;
  __shared__ __attribute__((aligned(16))) _Float16 As[2][TM * LDS_STRIDE];
  __shared__ __attribute__((aligned(16))) _Float16 Ws[2][TN * LDS_STRIDE];

  const int tid  = threadIdx.x;
  const int wave = tid >> 5;
  const int lane = tid & 31;
  const int wm   = wave & 3;      // 4 M-subblocks of 32
  const int wn   = wave >> 2;     // 2 N-subblocks of 64
  const int g    = lane >> 4;
  const int r16  = lane & 15;
  const int bm   = blockIdx.x * TM;
  const int bn   = blockIdx.y * TN;

  // loader mapping: thread t -> row t/2, col-half (t&1)*16
  const int la_row = tid >> 1;
  const int la_col = (tid & 1) << 4;

  long asrc;
  {
    int gm = bm + la_row;
    int b = gm / Lrow, l = gm % Lrow;
    int ls = l + shift;
    if (ls < 0) ls += Lrow;
    if (ls >= Lrow) ls -= Lrow;
    asrc = (long)(b * Lrow + ls) * (long)K + la_col;
  }
  const long wsrc = (long)(bn + la_row) * (long)K + la_col;
  const int ldsoff = la_row * LDS_STRIDE + la_col;

  // ---- prologue: stage tile 0 into LDS buffer 0 ----
  v8h aStage[2], wStage[2];
  load_cvt16(A + asrc, aStage);
  load_cvt16(W + wsrc, wStage);
  store16_lds(&As[0][ldsoff], aStage);
  store16_lds(&Ws[0][ldsoff], wStage);

  v8f acc[2][4] = {};
  const int nIter = K / TKK;

  for (int it = 0; it < nIter; ++it) {
    const int cur = it & 1;
    const bool hasNext = (it + 1 < nIter);
    if (hasNext) {
      // issue next tile's global loads early; they overlap this tile's WMMAs
      load_cvt16(A + asrc + (it + 1) * TKK, aStage);
      load_cvt16(W + wsrc + (it + 1) * TKK, wStage);
      if (it + 2 < nIter) {
        __builtin_prefetch(A + asrc + (it + 2) * TKK, 0, 3);
        __builtin_prefetch(W + wsrc + (it + 2) * TKK, 0, 3);
      }
    }
    __syncthreads();   // LDS writes for stage `it` visible; prior reads retired

    v16h afrag[2], bfrag[4];
    #pragma unroll
    for (int mi = 0; mi < 2; ++mi) {
      int row = wm * 32 + mi * 16 + r16;
      afrag[mi] = load_fragA(&As[cur][row * LDS_STRIDE], g);
    }
    #pragma unroll
    for (int ni = 0; ni < 4; ++ni) {
      int row = wn * 64 + ni * 16 + r16;
      bfrag[ni] = load_fragB(&Ws[cur][row * LDS_STRIDE], g);
    }
    #pragma unroll
    for (int mi = 0; mi < 2; ++mi)
      #pragma unroll
      for (int ni = 0; ni < 4; ++ni)
        acc[mi][ni] = __builtin_amdgcn_wmma_f32_16x16x32_f16(
            false, afrag[mi], false, bfrag[ni], (short)0, acc[mi][ni], false, false);

    if (hasNext) {
      // safe: every wave's reads of buffer cur^1 (iteration it-1) completed
      // before its WMMA dscnt-wait, which precedes this iteration's barrier
      store16_lds(&As[cur ^ 1][ldsoff], aStage);
      store16_lds(&Ws[cur ^ 1][ldsoff], wStage);
    }
  }

  // ---- epilogue: bias / accumulate / exact GELU / residual (branch-free) ----
  #pragma unroll
  for (int mi = 0; mi < 2; ++mi) {
    #pragma unroll
    for (int ni = 0; ni < 4; ++ni) {
      int n = bn + wn * 64 + ni * 16 + r16;
      #pragma unroll
      for (int r = 0; r < 8; ++r) {
        int m = bm + wm * 32 + mi * 16 + r + 8 * g;
        long idx = (long)m * N + n;
        float val = acc[mi][ni][r];
        if (flags & FLAG_ACC) val += C[idx];
        if (bias) val += bias[n];
        if (flags & FLAG_GELU) val = 0.5f * val * (1.0f + erff(val * 0.70710678f));
        if (resid) val += resid[idx];
        C[idx] = val;
      }
    }
  }
}

// ---------------------------------------------------------------------------
// Token embedding: circular conv1d (CIN=7, k=3) + sinusoidal pos-embed
// ---------------------------------------------------------------------------
__global__ __launch_bounds__(256)
void tok_embed(const float* __restrict__ xe, const float* __restrict__ tw,
               float* __restrict__ x)
{
  const int S = 2048;
  size_t i = (size_t)blockIdx.x * 256 + threadIdx.x;
  size_t tot = (size_t)8 * S * 512;
  if (i >= tot) return;
  int co = (int)(i & 511);
  size_t bl = i >> 9;
  int l = (int)(bl % S);
  int b = (int)(bl / S);
  float s = 0.f;
  #pragma unroll
  for (int t = 0; t < 3; ++t) {
    int ls = l + t - 1;
    if (ls < 0) ls += S;
    if (ls >= S) ls -= S;
    const float* xr = xe + ((size_t)(b * S + ls)) * 7;
    #pragma unroll
    for (int ci = 0; ci < 7; ++ci)
      s += xr[ci] * tw[(co * 7 + ci) * 3 + t];
  }
  int p = co >> 1;
  float freq = __expf(-(2.0f * (float)p) * (9.21034037f / 512.0f)); // ln(1e4)=9.2103
  float ang = (float)l * freq;
  s += (co & 1) ? cosf(ang) : sinf(ang);
  x[i] = s;
}

// ---------------------------------------------------------------------------
// ProbSparse attention helpers
// ---------------------------------------------------------------------------
__device__ inline unsigned hash_u32(unsigned x) {
  x ^= x >> 16; x *= 0x7feb352dU; x ^= x >> 15; x *= 0x846ca68bU; x ^= x >> 16;
  return x;
}

// one wave per (b,h,l): M[l] = max_j qk[l,j] - sum_j qk[l,j] / LK
__global__ __launch_bounds__(256)
void attn_sample(const float* __restrict__ q, const float* __restrict__ k,
                 float* __restrict__ Mout, int L, int U, unsigned seed)
{
  int gw = (blockIdx.x * 256 + threadIdx.x) >> 5;
  int lane = threadIdx.x & 31;
  int total = 64 * L; // B*H*L
  if (gw >= total) return;
  int l = gw % L;
  int bh = gw / L;
  int h = bh & 7, b = bh >> 3;
  const float* qr = q + ((size_t)(b * L + l) * 512 + h * 64);
  float q0 = qr[lane], q1 = qr[lane + 32];
  float mx = -3.4e38f, sm = 0.f;
  for (int j = 0; j < U; ++j) {
    unsigned idx = hash_u32((unsigned)(l * U + j) + seed) % (unsigned)L;
    const float* kr = k + ((size_t)(b * L + (int)idx) * 512 + h * 64);
    float p = q0 * kr[lane] + q1 * kr[lane + 32];
    #pragma unroll
    for (int o = 16; o > 0; o >>= 1) p += __shfl_xor(p, o, 32);
    mx = fmaxf(mx, p);
    sm += p;
  }
  if (lane == 0) Mout[(size_t)bh * L + l] = mx - sm / (float)L;
}

// block per (b,h): iterative top-u (without replacement, lowest index on ties)
__global__ __launch_bounds__(256)
void topk_kernel(const float* __restrict__ Msc, int* __restrict__ topi, int L, int u)
{
  __shared__ float sv[2048];
  __shared__ float rv[256];
  __shared__ int   ri[256];
  int bh = blockIdx.x, t = threadIdx.x;
  for (int i = t; i < 2048; i += 256)
    sv[i] = (i < L) ? Msc[(size_t)bh * L + i] : -3.4e38f;
  __syncthreads();
  for (int it = 0; it < u; ++it) {
    float bv = -3.4e38f; int bi = 0x7fffffff;
    for (int i = t; i < L; i += 256) {
      float v = sv[i];
      if (v > bv) { bv = v; bi = i; }
    }
    rv[t] = bv; ri[t] = bi;
    __syncthreads();
    for (int s = 128; s > 0; s >>= 1) {
      if (t < s) {
        if (rv[t + s] > rv[t] || (rv[t + s] == rv[t] && ri[t + s] < ri[t])) {
          rv[t] = rv[t + s]; ri[t] = ri[t + s];
        }
      }
      __syncthreads();
    }
    if (t == 0) { topi[bh * u + it] = ri[0]; sv[ri[0]] = -3.4e38f; }
    __syncthreads();
  }
}

// block per (b,h): vmean[bh,d] = mean_l v[b,h,l,d]
__global__ __launch_bounds__(256)
void vmean_kernel(const float* __restrict__ v, float* __restrict__ vmean, int L)
{
  __shared__ float red[256];
  int bh = blockIdx.x;
  int b = bh >> 3, h = bh & 7;
  int t = threadIdx.x;
  int d = t & 63, c = t >> 6;
  int chunk = L >> 2;
  float s = 0.f;
  for (int l = c * chunk; l < (c + 1) * chunk; ++l)
    s += v[(size_t)(b * L + l) * 512 + h * 64 + d];
  red[t] = s;
  __syncthreads();
  if (t < 64)
    vmean[bh * 64 + t] = (red[t] + red[t + 64] + red[t + 128] + red[t + 192]) / (float)L;
}

__global__ __launch_bounds__(256)
void ctx_fill(float* __restrict__ ctx, const float* __restrict__ vmean, int L)
{
  size_t i = (size_t)blockIdx.x * 256 + threadIdx.x;
  size_t tot = (size_t)8 * L * 512;
  if (i >= tot) return;
  int c = (int)(i & 511);
  int h = c >> 6, d = c & 63;
  size_t bl = i >> 9;
  int b = (int)(bl / L);
  ctx[i] = vmean[(b * 8 + h) * 64 + d];
}

// block per (b,h,ui): full softmax attention for one selected query row
__global__ __launch_bounds__(256)
void attn_update(const float* __restrict__ q, const float* __restrict__ k,
                 const float* __restrict__ v, const int* __restrict__ topi,
                 float* __restrict__ ctx, int L, int u)
{
  __shared__ float sc[2048];
  __shared__ float red[256];
  int bid = blockIdx.x;
  int ui = bid % u;
  int bh = bid / u;
  int h = bh & 7, b = bh >> 3;
  int li = topi[bh * u + ui];
  int t = threadIdx.x;
  const float* qr = q + ((size_t)(b * L + li) * 512 + h * 64);

  for (int kk = t; kk < L; kk += 256) {
    const float* kr = k + ((size_t)(b * L + kk) * 512 + h * 64);
    float s = 0.f;
    #pragma unroll 8
    for (int d = 0; d < 64; ++d) s += qr[d] * kr[d];
    sc[kk] = s * 0.125f; // / sqrt(64)
  }
  __syncthreads();
  float lm = -3.4e38f;
  for (int kk = t; kk < L; kk += 256) lm = fmaxf(lm, sc[kk]);
  red[t] = lm;
  __syncthreads();
  for (int s = 128; s > 0; s >>= 1) { if (t < s) red[t] = fmaxf(red[t], red[t + s]); __syncthreads(); }
  float mx = red[0];
  __syncthreads();
  float ls = 0.f;
  for (int kk = t; kk < L; kk += 256) { float e = __expf(sc[kk] - mx); sc[kk] = e; ls += e; }
  red[t] = ls;
  __syncthreads();
  for (int s = 128; s > 0; s >>= 1) { if (t < s) red[t] += red[t + s]; __syncthreads(); }
  float inv = 1.0f / red[0];
  __syncthreads();
  int d = t & 63, g4 = t >> 6;
  int chunk = L >> 2;
  float a = 0.f;
  for (int kk = g4 * chunk; kk < (g4 + 1) * chunk; ++kk)
    a += sc[kk] * v[(size_t)(b * L + kk) * 512 + h * 64 + d];
  red[t] = a;
  __syncthreads();
  if (t < 64)
    ctx[(size_t)(b * L + li) * 512 + h * 64 + t] =
        (red[t] + red[t + 64] + red[t + 128] + red[t + 192]) * inv;
}

// ---------------------------------------------------------------------------
// LayerNorm over last dim (512), one block per row
// ---------------------------------------------------------------------------
__global__ __launch_bounds__(256)
void layernorm_k(const float* __restrict__ x, float* __restrict__ y,
                 const float* __restrict__ g, const float* __restrict__ b, int rows)
{
  __shared__ float s1[256], s2[256];
  int r = blockIdx.x;
  if (r >= rows) return;
  int t = threadIdx.x;
  const float* xr = x + (size_t)r * 512;
  float v0 = xr[t], v1 = xr[t + 256];
  s1[t] = v0 + v1;
  s2[t] = v0 * v0 + v1 * v1;
  __syncthreads();
  for (int s = 128; s > 0; s >>= 1) { if (t < s) { s1[t] += s1[t + s]; s2[t] += s2[t + s]; } __syncthreads(); }
  float mean = s1[0] * (1.0f / 512.0f);
  float var = s2[0] * (1.0f / 512.0f) - mean * mean;
  float inv = rsqrtf(var + 1e-5f);
  float* yr = y + (size_t)r * 512;
  yr[t]       = (v0 - mean) * inv * g[t]       + b[t];
  yr[t + 256] = (v1 - mean) * inv * g[t + 256] + b[t + 256];
}

// ---------------------------------------------------------------------------
// Distil: repack conv weights, batchnorm stats, fused BN+ELU+maxpool(3,s2,p1)
// ---------------------------------------------------------------------------
__global__ __launch_bounds__(256)
void repack_dconv(const float* __restrict__ src, float* __restrict__ dst)
{
  int i = blockIdx.x * 256 + threadIdx.x;
  if (i >= 3 * 512 * 512) return;
  int t = i / (512 * 512);
  int r = i - t * 512 * 512;
  int co = r >> 9, ci = r & 511;
  dst[t * 512 * 512 + co * 512 + ci] = src[(co * 512 + ci) * 3 + t];
}

__global__ __launch_bounds__(256)
void bnstats_k(const float* __restrict__ y, float* __restrict__ mu,
               float* __restrict__ inv, int rows)
{
  __shared__ float s1[256], s2[256];
  int c = blockIdx.x, t = threadIdx.x;
  float s = 0.f, ss = 0.f;
  for (int r = t; r < rows; r += 256) {
    float v = y[(size_t)r * 512 + c];
    s += v; ss += v * v;
  }
  s1[t] = s; s2[t] = ss;
  __syncthreads();
  for (int st = 128; st > 0; st >>= 1) { if (t < st) { s1[t] += s1[t + st]; s2[t] += s2[t + st]; } __syncthreads(); }
  if (t == 0) {
    float m = s1[0] / (float)rows;
    float var = s2[0] / (float)rows - m * m;
    mu[c] = m;
    inv[c] = rsqrtf(var + 1e-5f);
  }
}

__global__ __launch_bounds__(256)
void distil_pool(const float* __restrict__ y, float* __restrict__ out,
                 const float* __restrict__ mu, const float* __restrict__ inv,
                 const float* __restrict__ g, const float* __restrict__ be, int L)
{
  int Lo = L >> 1;
  size_t i = (size_t)blockIdx.x * 256 + threadIdx.x;
  size_t tot = (size_t)8 * Lo * 512;
  if (i >= tot) return;
  int c = (int)(i & 511);
  size_t bl = i >> 9;
  int lo = (int)(bl % Lo);
  int b = (int)(bl / Lo);
  float m = mu[c], iv = inv[c], gg = g[c], bb = be[c];
  float best = -3.4e38f;
  #pragma unroll
  for (int tdx = -1; tdx <= 1; ++tdx) {
    int l = 2 * lo + tdx;
    if (l < 0 || l >= L) continue;
    float v = (y[(size_t)(b * L + l) * 512 + c] - m) * iv * gg + bb;
    v = (v > 0.f) ? v : (__expf(v) - 1.f);  // ELU
    best = fmaxf(best, v);
  }
  out[i] = best;
}

// ---------------------------------------------------------------------------
// Final head: out[r, o] = x[r, :] . end_w[o, :] + end_b[o]   (N=7, tiny)
// ---------------------------------------------------------------------------
__global__ __launch_bounds__(256)
void head_k(const float* __restrict__ x, const float* __restrict__ w,
            const float* __restrict__ b, float* __restrict__ out, int rows)
{
  int i = blockIdx.x * 256 + threadIdx.x;
  if (i >= rows * 7) return;
  int o = i % 7, r = i / 7;
  const float* xr = x + (size_t)r * 512;
  const float* wr = w + o * 512;
  float s = 0.f;
  #pragma unroll 8
  for (int d = 0; d < 512; ++d) s += xr[d] * wr[d];
  out[i] = s + b[o];
}

// ---------------------------------------------------------------------------
// Host orchestration
// ---------------------------------------------------------------------------
static inline void gemm_launch(hipStream_t s, const float* A, const float* W,
                               const float* bias, const float* resid, float* C,
                               int M, int N, int K, int Lrow, int shift, int flags)
{
  dim3 grid((M + TM - 1) / TM, (N + TN - 1) / TN);
  gemm_f16wmma<<<grid, 256, 0, s>>>(A, W, bias, resid, C, M, N, K, Lrow, shift, flags);
}

extern "C" void kernel_launch(void* const* d_in, const int* in_sizes, int n_in,
                              void* d_out, int out_size, void* d_ws, size_t ws_size,
                              hipStream_t stream) {
  (void)in_sizes; (void)n_in; (void)out_size; (void)ws_size;
  const float* xe      = (const float*)d_in[0];
  const float* tok_w   = (const float*)d_in[1];
  const float* Wq      = (const float*)d_in[2];
  const float* bq      = (const float*)d_in[3];
  const float* Wk      = (const float*)d_in[4];
  const float* bk      = (const float*)d_in[5];
  const float* Wv      = (const float*)d_in[6];
  const float* bv      = (const float*)d_in[7];
  const float* Wo      = (const float*)d_in[8];
  const float* bo      = (const float*)d_in[9];
  const float* conv1_w = (const float*)d_in[10];
  const float* conv1_b = (const float*)d_in[11];
  const float* conv2_w = (const float*)d_in[12];
  const float* conv2_b = (const float*)d_in[13];
  const float* ln1_g   = (const float*)d_in[14];
  const float* ln1_b   = (const float*)d_in[15];
  const float* ln2_g   = (const float*)d_in[16];
  const float* ln2_b   = (const float*)d_in[17];
  const float* dconv_w = (const float*)d_in[18];
  const float* dconv_b = (const float*)d_in[19];
  const float* bn_g    = (const float*)d_in[20];
  const float* bn_b    = (const float*)d_in[21];
  const float* lnf_g   = (const float*)d_in[22];
  const float* lnf_b   = (const float*)d_in[23];
  const float* end_w   = (const float*)d_in[24];
  const float* end_b   = (const float*)d_in[25];
  float* out = (float*)d_out;

  float* ws = (float*)d_ws;
  const size_t SZ = (size_t)16384 * 512;   // 8,388,608 floats
  float* bufA  = ws;
  float* bufB  = ws + SZ;
  float* qbuf  = ws + 2 * SZ;
  float* kbuf  = ws + 3 * SZ;
  float* vbuf  = ws + 4 * SZ;
  float* bufC  = ws + 5 * SZ;
  float* y1    = ws + 6 * SZ;              // 4096 x 2048 FFN chunk == SZ
  float* Msc   = ws + 7 * SZ;              // 131072
  float* vmean = Msc + 131072;             // 4096
  int*   topi  = (int*)(vmean + 4096);     // <= 1536 ints (2048 slots reserved)
  float* wconv = vmean + 4096 + 2048;      // 3*512*512
  float* bnmu  = wconv + 3 * 512 * 512;    // 512
  float* bninv = bnmu + 512;               // 512

  // ---- token embedding + positional encoding ----
  {
    size_t tot = (size_t)8 * 2048 * 512;
    tok_embed<<<(unsigned)((tot + 255) / 256), 256, 0, stream>>>(xe, tok_w, bufA);
  }

  const int uarr[3] = {24, 21, 21};  // 3*ceil(ln L) for L = 2048/1024/512
  int L = 2048;

  for (int l = 0; l < 3; ++l) {
    const int Mrows = 8 * L;
    const float* Wq_l = Wq + (size_t)l * 512 * 512;
    const float* Wk_l = Wk + (size_t)l * 512 * 512;
    const float* Wv_l = Wv + (size_t)l * 512 * 512;
    const float* Wo_l = Wo + (size_t)l * 512 * 512;

    // QKV projections
    gemm_launch(stream, bufA, Wq_l, bq + l * 512, nullptr, qbuf, Mrows, 512, 512, Mrows, 0, 0);
    gemm_launch(stream, bufA, Wk_l, bk + l * 512, nullptr, kbuf, Mrows, 512, 512, Mrows, 0, 0);
    gemm_launch(stream, bufA, Wv_l, bv + l * 512, nullptr, vbuf, Mrows, 512, 512, Mrows, 0, 0);

    // ProbSparse attention
    const int U = uarr[l];
    {
      int waves = 64 * L;
      int blocks = (waves * 32 + 255) / 256;
      attn_sample<<<blocks, 256, 0, stream>>>(qbuf, kbuf, Msc, L, U, 42u + (unsigned)l * 1337u);
    }
    topk_kernel<<<64, 256, 0, stream>>>(Msc, topi, L, U);
    vmean_kernel<<<64, 256, 0, stream>>>(vbuf, vmean, L);
    {
      size_t tot = (size_t)8 * L * 512;
      ctx_fill<<<(unsigned)((tot + 255) / 256), 256, 0, stream>>>(bufC, vmean, L);
    }
    attn_update<<<64 * U, 256, 0, stream>>>(qbuf, kbuf, vbuf, topi, bufC, L, U);

    // output projection + residual, then LN1
    gemm_launch(stream, bufC, Wo_l, bo + l * 512, bufA, bufB, Mrows, 512, 512, Mrows, 0, 0);
    layernorm_k<<<Mrows, 256, 0, stream>>>(bufB, bufA, ln1_g + l * 512, ln1_b + l * 512, Mrows);

    // FFN (chunked to 4096 rows so the 2048-wide intermediate reuses one buffer)
    {
      const int CH = 4096;
      int nch = Mrows / CH;
      const float* w1_l = conv1_w + (size_t)l * 2048 * 512;
      const float* w2_l = conv2_w + (size_t)l * 512 * 2048;
      for (int c = 0; c < nch; ++c) {
        const float* ap = bufA + (size_t)c * CH * 512;
        float* cp = bufB + (size_t)c * CH * 512;
        gemm_launch(stream, ap, w1_l, conv1_b + l * 2048, nullptr, y1, CH, 2048, 512, CH, 0, FLAG_GELU);
        gemm_launch(stream, y1, w2_l, conv2_b + l * 512, ap, cp, CH, 512, 2048, CH, 0, 0);
      }
    }
    layernorm_k<<<Mrows, 256, 0, stream>>>(bufB, bufA, ln2_g + l * 512, ln2_b + l * 512, Mrows);

    // distilling layer (circular conv as 3 shifted GEMMs, BN, ELU, maxpool s2)
    if (l < 2) {
      repack_dconv<<<(3 * 512 * 512 + 255) / 256, 256, 0, stream>>>(
          dconv_w + (size_t)l * 512 * 512 * 3, wconv);
      for (int t = 0; t < 3; ++t) {
        gemm_launch(stream, bufA, wconv + (size_t)t * 512 * 512,
                    (t == 0) ? (dconv_b + l * 512) : nullptr, nullptr,
                    bufB, Mrows, 512, 512, L, t - 1, (t == 0) ? 0 : FLAG_ACC);
      }
      bnstats_k<<<512, 256, 0, stream>>>(bufB, bnmu, bninv, Mrows);
      {
        size_t tot = (size_t)8 * (L / 2) * 512;
        distil_pool<<<(unsigned)((tot + 255) / 256), 256, 0, stream>>>(
            bufB, bufA, bnmu, bninv, bn_g + l * 512, bn_b + l * 512, L);
      }
      L >>= 1;
    }
  }

  // final LN + head (B=8, L=512 -> 4096 rows, 7 outputs)
  layernorm_k<<<8 * 512, 256, 0, stream>>>(bufA, bufB, lnf_g, lnf_b, 8 * 512);
  head_k<<<(8 * 512 * 7 + 255) / 256, 256, 0, stream>>>(bufB, end_w, end_b, out, 8 * 512);
}
